// MultiHeadAttention_62294205661880
// MI455X (gfx1250) — compile-verified, matrix operand order checked by reading, not run on hardware
//
#include <hip/hip_runtime.h>
#include <hip/hip_bf16.h>
#include <math.h>

typedef _Float16 f16;
typedef __attribute__((ext_vector_type(16))) _Float16 v16h;
typedef __attribute__((ext_vector_type(8)))  _Float16 v8h;
typedef __attribute__((ext_vector_type(8)))  float    v8f;
typedef __attribute__((ext_vector_type(4)))  int      v4i;

union V16 { v16h v; v8h h[2]; };

constexpr int BATCH = 2, SEQ = 2048, DM = 2048, NH = 16, DH = 128;
constexpr int TD    = 3 * DM;            // 6144
constexpr int MROWS = BATCH * SEQ;       // 4096
constexpr int NWQ   = TD * DM;           // 12582912
constexpr int NWO   = DM * DM;           // 4194304
constexpr int NX    = MROWS * DM;        // 8388608

// ---- workspace layout (bytes) ----
constexpr size_t OFF_RED = 0;                                  // 16 KB reduction area
constexpr size_t OFF_XH  = 16384;                              // x f16 (later aliased by qh)
constexpr size_t OFF_WQT = OFF_XH  + (size_t)NX * 2;           // ternary w_qkv f16
constexpr size_t OFF_WOT = OFF_WQT + (size_t)NWQ * 2;          // ternary w_out f16
constexpr size_t OFF_QKV = OFF_WOT + (size_t)NWO * 2;          // qkv f16 (later aliased by attn out)
constexpr size_t OFF_KH  = OFF_QKV + (size_t)MROWS * TD * 2;   // k rotated f16
constexpr size_t OFF_VH  = OFF_KH  + (size_t)NX * 2;           // v f16
// total ≈ 134.3 MB

// ---- CDNA5 async global->LDS path (guarded; falls back to sync copy) ----
#if __has_builtin(__builtin_amdgcn_global_load_async_to_lds_b128)
#define HAS_ASYNC_LDS 1
#else
#define HAS_ASYNC_LDS 0
#endif
#if __has_builtin(__builtin_amdgcn_s_wait_asynccnt)
#define HAS_WAIT_ASYNC 1
#else
#define HAS_WAIT_ASYNC 0
#endif

#if HAS_ASYNC_LDS
typedef __attribute__((address_space(1))) v4i g_v4i;
typedef __attribute__((address_space(3))) v4i l_v4i;
#endif

static __device__ inline void copy16_async(const f16* g, f16* l) {
#if HAS_ASYNC_LDS
  __builtin_amdgcn_global_load_async_to_lds_b128((g_v4i*)g, (l_v4i*)l, 0, 0);
#else
  *(v8h*)l = *(const v8h*)g;
#endif
}
static __device__ inline void wait_async() {
#if HAS_ASYNC_LDS
#if HAS_WAIT_ASYNC
  __builtin_amdgcn_s_wait_asynccnt(0);
#else
  asm volatile("s_wait_asynccnt 0" ::: "memory");
#endif
#endif
}

static __device__ inline v8f zero8f() {
  v8f z = {0.f,0.f,0.f,0.f,0.f,0.f,0.f,0.f};
  return z;
}
static __device__ inline v16h load_frag(const f16* p0, const f16* p1) {
  V16 t; t.h[0] = *(const v8h*)p0; t.h[1] = *(const v8h*)p1; return t.v;
}

// ---------------- scale reduction ----------------
__global__ __launch_bounds__(256) void absum_partial(const float* __restrict__ w,
                                                     float* __restrict__ part, int n) {
  __shared__ float sm[256];
  float s = 0.f;
  for (int i = blockIdx.x * blockDim.x + threadIdx.x; i < n; i += gridDim.x * blockDim.x)
    s += fabsf(w[i]);
  sm[threadIdx.x] = s; __syncthreads();
  for (int off = 128; off; off >>= 1) {
    if ((int)threadIdx.x < off) sm[threadIdx.x] += sm[threadIdx.x + off];
    __syncthreads();
  }
  if (threadIdx.x == 0) part[blockIdx.x] = sm[0];
}

__global__ __launch_bounds__(256) void finalize_scales(float* __restrict__ red) {
  __shared__ float sm[256];
  float s1 = 0.f, s2 = 0.f;
  for (int i = threadIdx.x; i < 1024; i += 256) {
    s1 += red[16 + i];
    s2 += red[16 + 1024 + i];
  }
  sm[threadIdx.x] = s1; __syncthreads();
  for (int off = 128; off; off >>= 1) {
    if ((int)threadIdx.x < off) sm[threadIdx.x] += sm[threadIdx.x + off];
    __syncthreads();
  }
  s1 = sm[0]; __syncthreads();
  sm[threadIdx.x] = s2; __syncthreads();
  for (int off = 128; off; off >>= 1) {
    if ((int)threadIdx.x < off) sm[threadIdx.x] += sm[threadIdx.x + off];
    __syncthreads();
  }
  s2 = sm[0];
  if (threadIdx.x == 0) {
    red[0] = fmaxf(s1 / (float)NWQ, 1e-5f);
    red[1] = fmaxf(s2 / (float)NWO, 1e-5f);
  }
}

// ---------------- quantize / convert ----------------
__global__ __launch_bounds__(256) void quant_ternary(const float* __restrict__ w,
                                                     const float* __restrict__ red, int sidx,
                                                     f16* __restrict__ out) {
  int i = blockIdx.x * 256 + threadIdx.x;
  float inv = 1.f / red[sidx];
  float q = rintf(w[i] * inv);            // round-half-even, matches jnp.round
  out[i] = (f16)fminf(1.f, fmaxf(-1.f, q));
}

__global__ __launch_bounds__(256) void cvt_f16(const float* __restrict__ x, f16* __restrict__ out) {
  int i = blockIdx.x * 256 + threadIdx.x;
  out[i] = (f16)x[i];
}

// ---------------- ternary GEMM: C[M,N] = A[M,K] * Bt[N,K]^T * scale ----------------
// Double-buffered LDS, async global->LDS staging overlapped with WMMA.
constexpr int BM = 128, BN = 128, BK = 32;

template <bool OUT_F16>
__global__ __launch_bounds__(256) void gemm_ternary(const f16* __restrict__ A,
                                                    const f16* __restrict__ Bt,
                                                    void* __restrict__ Cout,
                                                    const float* __restrict__ red, int sidx,
                                                    int M, int N, int K) {
  __shared__ f16 As[2][BM][BK + 8];
  __shared__ f16 Bs[2][BM][BK + 8];

  const int tid  = threadIdx.x;
  const int lane = tid & 31, wave = tid >> 5;     // 8 waves
  const int waveM = wave >> 1, waveN = wave & 1;  // 4x2 wave grid, each 32x64
  const int rowBase = blockIdx.y * BM;
  const int colBase = blockIdx.x * BN;
  const float scale = red[sidx];

  const int lcol = (tid & 3) * 8;   // 8-f16 chunk within 32-wide K slab
  const int lrow = tid >> 2;        // 64 rows per pass, 2 passes

  const int c0   = (lane < 16) ? 0 : 8;  // fragment K-base per lane half
  const int frow = lane & 15;

  auto stage = [&](int buf, int k0) {
#pragma unroll
    for (int i = 0; i < 2; ++i) {
      int r = lrow + i * 64;
      copy16_async(&A[(size_t)(rowBase + r) * K + k0 + lcol], &As[buf][r][lcol]);
      copy16_async(&Bt[(size_t)(colBase + r) * K + k0 + lcol], &Bs[buf][r][lcol]);
    }
  };

  v8f acc[2][4];
#pragma unroll
  for (int i = 0; i < 2; ++i)
#pragma unroll
    for (int j = 0; j < 4; ++j) acc[i][j] = zero8f();

  stage(0, 0);
  wait_async();
  __syncthreads();

  const int nk = K / BK;
  for (int t = 0; t < nk; ++t) {
    const int cur = t & 1;
    if (t + 1 < nk) stage(cur ^ 1, (t + 1) * BK);   // prefetch next slab (async)

    v16h af[2], bf[4];
#pragma unroll
    for (int mt = 0; mt < 2; ++mt) {
      int r = waveM * 32 + mt * 16 + frow;
      af[mt] = load_frag(&As[cur][r][c0], &As[cur][r][c0 + 16]);
    }
#pragma unroll
    for (int nt = 0; nt < 4; ++nt) {
      int c = waveN * 64 + nt * 16 + frow;
      bf[nt] = load_frag(&Bs[cur][c][c0], &Bs[cur][c][c0 + 16]);
    }
#pragma unroll
    for (int mt = 0; mt < 2; ++mt)
#pragma unroll
      for (int nt = 0; nt < 4; ++nt)
        acc[mt][nt] = __builtin_amdgcn_wmma_f32_16x16x32_f16(
            false, af[mt], false, bf[nt], (short)0, acc[mt][nt], false, false);

    if (t + 1 < nk) wait_async();   // own async loads landed; barrier makes it collective
    __syncthreads();
  }

  const int colIn = lane & 15;
  const int rowHalf = (lane >> 4) * 8;
#pragma unroll
  for (int mt = 0; mt < 2; ++mt)
#pragma unroll
    for (int nt = 0; nt < 4; ++nt)
#pragma unroll
      for (int r = 0; r < 8; ++r) {
        int grow = rowBase + waveM * 32 + mt * 16 + rowHalf + r;
        int gcol = colBase + waveN * 64 + nt * 16 + colIn;
        float v = acc[mt][nt][r] * scale;
        if (OUT_F16) ((f16*)Cout)[(size_t)grow * N + gcol] = (f16)v;
        else         ((float*)Cout)[(size_t)grow * N + gcol] = v;
      }
}

// ---------------- RoPE + head split: qkv[B,S,3,H,Dh] -> q/k/v [B,H,S,Dh] ----------------
__global__ __launch_bounds__(256) void rope_split(const f16* __restrict__ qkv,
                                                  f16* __restrict__ qh, f16* __restrict__ kh,
                                                  f16* __restrict__ vh) {
  int idx = blockIdx.x * 256 + threadIdx.x;   // B*S*H*(Dh/2) threads
  int i = idx & 63;
  int h = (idx >> 6) & 15;
  int s = (idx >> 10) & (SEQ - 1);
  int b = idx >> 21;

  float inv_freq = powf(10000.f, -(float)i / 64.f);
  float ang = (float)s * inv_freq;
  float c = cosf(ang), sn = sinf(ang);

  size_t base = (((size_t)(b * SEQ + s) * 3) * NH + h) * DH;   // t=0 (q)
  size_t obase = (((size_t)(b * NH + h)) * SEQ + s) * DH;

  float q1 = (float)qkv[base + i],                   q2 = (float)qkv[base + i + 64];
  float k1 = (float)qkv[base + (size_t)NH * DH + i], k2 = (float)qkv[base + (size_t)NH * DH + i + 64];
  qh[obase + i]      = (f16)(q1 * c - q2 * sn);
  qh[obase + i + 64] = (f16)(q1 * sn + q2 * c);
  kh[obase + i]      = (f16)(k1 * c - k2 * sn);
  kh[obase + i + 64] = (f16)(k1 * sn + k2 * c);
  vh[obase + i]      = qkv[base + 2 * (size_t)NH * DH + i];
  vh[obase + i + 64] = qkv[base + 2 * (size_t)NH * DH + i + 64];
}

// ---------------- flash attention: one block = (b,h, 128 q-rows) ----------------
constexpr int QB = 128;   // q rows per block (8 waves x 16)
constexpr int KB = 64;    // keys per iteration

__global__ __launch_bounds__(256) void flash_attn(const f16* __restrict__ qh,
                                                  const f16* __restrict__ kh,
                                                  const f16* __restrict__ vh,
                                                  f16* __restrict__ oh) {
  __shared__ f16 Ks[KB][DH + 8];        // keys row-major
  __shared__ f16 Vt[DH][KB + 8];        // V transposed: [dh][key]
  __shared__ f16 Ps[8][16][KB + 8];     // per-wave P tile (C-layout -> A-layout bridge)

  const int bh = blockIdx.y;            // b*NH + h
  const int b  = bh >> 4, h = bh & 15;
  const int qbase = blockIdx.x * QB;

  const f16* Q  = qh + (size_t)bh * SEQ * DH;
  const f16* Kp = kh + (size_t)bh * SEQ * DH;
  const f16* Vp = vh + (size_t)bh * SEQ * DH;

  const int tid = threadIdx.x, lane = tid & 31, wave = tid >> 5;
  const int c0 = (lane < 16) ? 0 : 8;
  const int lrow = lane & 15;
  const int rowHalf = (lane >> 4) * 8;

  // Q fragments for this wave's 16 rows: 4 x (16x32) over Dh=128
  const int qrow = qbase + wave * 16 + lrow;
  v16h qf[4];
#pragma unroll
  for (int kk = 0; kk < 4; ++kk) {
    const f16* p = &Q[(size_t)qrow * DH + kk * 32];
    qf[kk] = load_frag(p + c0, p + c0 + 16);
  }

  float m_prev[8], l_sum[8];
  v8f oacc[8];
#pragma unroll
  for (int r = 0; r < 8; ++r) { m_prev[r] = -1e30f; l_sum[r] = 0.f; }
#pragma unroll
  for (int td = 0; td < 8; ++td) oacc[td] = zero8f();

  const float sscale = 0.08838834764831845f;   // 1/sqrt(128)
  const int nkb = qbase / KB + QB / KB;        // causal: keys up to qbase+QB-1

  for (int jb = 0; jb < nkb; ++jb) {
    const int kb0 = jb * KB;
    __syncthreads();
    // cooperative load: K rows (async direct-to-LDS) + V transposed (VGPR bounce)
#pragma unroll
    for (int i = 0; i < 4; ++i) {
      int r = (tid >> 4) + i * 16;            // 0..63
      int cg = (tid & 15) * 8;                // 0..120
      copy16_async(&Kp[(size_t)(kb0 + r) * DH + cg], &Ks[r][cg]);
      v8h vv = *(const v8h*)&Vp[(size_t)(kb0 + r) * DH + cg];
#pragma unroll
      for (int e = 0; e < 8; ++e) Vt[cg + e][r] = vv[e];
    }
    wait_async();
    __syncthreads();

    // scores: 4 tiles of 16 keys, K-loop over Dh
    v8f sacc[4];
#pragma unroll
    for (int nt = 0; nt < 4; ++nt) {
      sacc[nt] = zero8f();
#pragma unroll
      for (int kk = 0; kk < 4; ++kk) {
        const f16* p = &Ks[nt * 16 + lrow][kk * 32];
        v16h bf = load_frag(p + c0, p + c0 + 16);
        sacc[nt] = __builtin_amdgcn_wmma_f32_16x16x32_f16(
            false, qf[kk], false, bf, (short)0, sacc[nt], false, false);
      }
    }

    // online softmax per row (8 rows per lane-half; 16-lane shfl reductions)
#pragma unroll
    for (int r = 0; r < 8; ++r) {
      const int myrow = qbase + wave * 16 + rowHalf + r;
      float mx = -1e30f;
#pragma unroll
      for (int nt = 0; nt < 4; ++nt) {
        int key = kb0 + nt * 16 + (lane & 15);
        float s = sacc[nt][r] * sscale;
        if (key > myrow) s = -1e30f;          // causal mask
        sacc[nt][r] = s;
        mx = fmaxf(mx, s);
      }
#pragma unroll
      for (int off = 1; off < 16; off <<= 1) mx = fmaxf(mx, __shfl_xor(mx, off, 32));
      float mnew = fmaxf(m_prev[r], mx);
      float alpha = __expf(m_prev[r] - mnew);
      float rsum = 0.f;
#pragma unroll
      for (int nt = 0; nt < 4; ++nt) {
        float p = __expf(sacc[nt][r] - mnew);
        sacc[nt][r] = p;
        rsum += p;
      }
#pragma unroll
      for (int off = 1; off < 16; off <<= 1) rsum += __shfl_xor(rsum, off, 32);
      l_sum[r] = l_sum[r] * alpha + rsum;
      m_prev[r] = mnew;
#pragma unroll
      for (int td = 0; td < 8; ++td) oacc[td][r] *= alpha;
      // stash P in per-wave LDS slab (C-layout scatter)
#pragma unroll
      for (int nt = 0; nt < 4; ++nt)
        Ps[wave][rowHalf + r][nt * 16 + (lane & 15)] = (f16)sacc[nt][r];
    }

    asm volatile("s_wait_dscnt 0" ::: "memory");   // in-wave LDS RAW fence

    // reload P as A-fragments: 16 x 64 -> two 16x32 frags
    v16h pf[2];
#pragma unroll
    for (int kk = 0; kk < 2; ++kk) {
      const f16* p = &Ps[wave][lrow][kk * 32];
      pf[kk] = load_frag(p + c0, p + c0 + 16);
    }
    // O += P * V  (8 dh-tiles of 16)
#pragma unroll
    for (int td = 0; td < 8; ++td) {
#pragma unroll
      for (int kk = 0; kk < 2; ++kk) {
        const f16* p = &Vt[td * 16 + lrow][kk * 32];
        v16h vf = load_frag(p + c0, p + c0 + 16);
        oacc[td] = __builtin_amdgcn_wmma_f32_16x16x32_f16(
            false, pf[kk], false, vf, (short)0, oacc[td], false, false);
      }
    }
  }

  // epilogue: divide by row sum, write [B,S,H*Dh]
#pragma unroll
  for (int td = 0; td < 8; ++td)
#pragma unroll
    for (int r = 0; r < 8; ++r) {
      int srow = qbase + wave * 16 + rowHalf + r;
      int dh = td * 16 + (lane & 15);
      float v = oacc[td][r] / l_sum[r];
      oh[((size_t)b * SEQ + srow) * DM + h * DH + dh] = (f16)v;
    }
}

// ---------------- host ----------------
extern "C" void kernel_launch(void* const* d_in, const int* in_sizes, int n_in,
                              void* d_out, int out_size, void* d_ws, size_t ws_size,
                              hipStream_t stream) {
  const float* x    = (const float*)d_in[0];
  const float* wqkv = (const float*)d_in[1];
  const float* wout = (const float*)d_in[2];
  float* out = (float*)d_out;

  char* ws = (char*)d_ws;
  float* red = (float*)(ws + OFF_RED);
  f16* xh  = (f16*)(ws + OFF_XH);
  f16* wqt = (f16*)(ws + OFF_WQT);
  f16* wot = (f16*)(ws + OFF_WOT);
  f16* qkv = (f16*)(ws + OFF_QKV);
  f16* kh  = (f16*)(ws + OFF_KH);
  f16* vh  = (f16*)(ws + OFF_VH);
  f16* qh  = xh;    // alias: x dead after gemm1
  f16* oh  = qkv;   // alias: qkv dead after rope

  absum_partial<<<1024, 256, 0, stream>>>(wqkv, red + 16, NWQ);
  absum_partial<<<1024, 256, 0, stream>>>(wout, red + 16 + 1024, NWO);
  finalize_scales<<<1, 256, 0, stream>>>(red);

  quant_ternary<<<NWQ / 256, 256, 0, stream>>>(wqkv, red, 0, wqt);
  quant_ternary<<<NWO / 256, 256, 0, stream>>>(wout, red, 1, wot);
  cvt_f16<<<NX / 256, 256, 0, stream>>>(x, xh);

  gemm_ternary<true><<<dim3(TD / BN, MROWS / BM), 256, 0, stream>>>(
      xh, wqt, (void*)qkv, red, 0, MROWS, TD, DM);

  rope_split<<<(BATCH * SEQ * NH * (DH / 2)) / 256, 256, 0, stream>>>(qkv, qh, kh, vh);

  flash_attn<<<dim3(SEQ / QB, BATCH * NH), 256, 0, stream>>>(qh, kh, vh, oh);

  gemm_ternary<false><<<dim3(DM / BN, MROWS / BM), 256, 0, stream>>>(
      oh, wot, (void*)out, red, 1, MROWS, DM, DM);
}
